// axs_89807766159735
// MI455X (gfx1250) — matrix operand order repaired and checked by reading, not compiled, not run on hardware
//
#include <hip/hip_runtime.h>

#define N28   28
#define NN    784     // 28*28
#define KPAD  800     // 25 * 32, zero-padded K for WMMA
#define BROWS 65536

typedef __attribute__((ext_vector_type(16))) __bf16 v16bf;
typedef __attribute__((ext_vector_type(8)))  float  v8f;
typedef __attribute__((ext_vector_type(4)))  float  f32x4;

// ---------------------------------------------------------------------------
// Kernel 1: materialize W in bf16, row-major with rows padded to KPAD=800.
// W[i, j] = exp(min(ep[j],0) * d2(i,j)) * (d2 < 9), where i = grid index
// (a,b) with grid=(x=b, y=a) per meshgrid(indexing='xy'), j = pos2d index.
// ---------------------------------------------------------------------------
__global__ void axs_build_w(const float* __restrict__ pos2d,
                            const float* __restrict__ ep,
                            __bf16* __restrict__ Wbf) {
    int idx = blockIdx.x * blockDim.x + threadIdx.x;
    if (idx >= NN * KPAD) return;
    int i = idx / KPAD;            // output (grid) flat index
    int j = idx - i * KPAD;        // input (pos) flat index, padded
    float w = 0.0f;
    if (j < NN) {
        int a = i / N28;
        int b = i - a * N28;
        float px = pos2d[2 * j + 0];
        float py = pos2d[2 * j + 1];
        float dx = (float)b - px;
        float dy = (float)a - py;
        float d2 = dx * dx + dy * dy;
        float e  = ep[j];
        float en = e < 0.0f ? e : 0.0f;   // -relu(-ep) == min(ep, 0)
        w = (d2 < 9.0f) ? __expf(en * d2) : 0.0f;
    }
    Wbf[(size_t)i * KPAD + j] = (__bf16)w;
}

// ---------------------------------------------------------------------------
// Kernel 2: out = (x @ W^T) * weight  via v_wmma_f32_16x16x32_bf16.
// Block = 256 threads = 8 waves; block covers 128 rows; each wave covers a
// 16-row strip and sweeps all 49 column tiles in 7 groups of 7 accumulators.
// Per K-slab: all 7 B fragments are fetched into distinct registers first so
// the 14 b128 loads stay in flight while the WMMAs drain with partial waits.
// ---------------------------------------------------------------------------
__global__ __launch_bounds__(256) void axs_gemm_wmma(
        const float*  __restrict__ x,
        const __bf16* __restrict__ Wbf,
        const float*  __restrict__ weight,
        float*        __restrict__ out) {
    const int lane    = threadIdx.x & 31;
    const int wave    = threadIdx.x >> 5;        // 0..7
    const int hi      = lane >> 4;               // 0 (lanes 0-15) / 1 (16-31)
    const int l15     = lane & 15;
    const int rowBase = blockIdx.x * 128 + wave * 16;

    const float* xrow = x + (size_t)(rowBase + l15) * NN;  // A: row = lane%16

    const v8f   vzero = {0.f, 0.f, 0.f, 0.f, 0.f, 0.f, 0.f, 0.f};
    const f32x4 qzero = {0.f, 0.f, 0.f, 0.f};

#pragma unroll 1
    for (int g = 0; g < 7; ++g) {
        // Base of this group's B columns; per-tile delta is a constant
        // 16*KPAD elements, so it folds into immediate offsets.
        const __bf16* wbase = Wbf + (size_t)(g * 7 * 16 + l15) * KPAD;

        v8f acc[7];
#pragma unroll
        for (int t = 0; t < 7; ++t) acc[t] = vzero;

#pragma unroll 1
        for (int kk = 0; kk < 25; ++kk) {
            const int kb = kk * 32;

            // ---- A fragment (16x32 bf16): ISA layout
            //   lanes 0-15 : elems 0..7 = K kb+0..7,  elems 8..15 = K kb+16..23
            //   lanes 16-31: elems 0..7 = K kb+8..15, elems 8..15 = K kb+24..31
            f32x4 f0 = *(const f32x4*)(xrow + kb + hi * 8);
            f32x4 f1 = *(const f32x4*)(xrow + kb + hi * 8 + 4);
            f32x4 f2, f3;
            if (kk < 24) {
                f2 = *(const f32x4*)(xrow + kb + 16 + hi * 8);
                f3 = *(const f32x4*)(xrow + kb + 16 + hi * 8 + 4);
            } else {                    // K 784..799 padded with zeros
                f2 = qzero;
                f3 = qzero;
            }

            // ---- B fragments (32x16 bf16): lane = column, same K order as A
            //   elems 0..7 = K kb+8*hi+0..7, elems 8..15 = K kb+16+8*hi+0..7.
            //   Distinct destination registers -> loads overlap the WMMAs.
            v16bf bfr[7];
#pragma unroll
            for (int t = 0; t < 7; ++t) {
                typedef __attribute__((ext_vector_type(8))) __bf16 v8bf_t;
                const __bf16* bp = wbase + (size_t)t * 16 * KPAD + kb + hi * 8;
                bfr[t] = __builtin_shufflevector(*(const v8bf_t*)bp, *(const v8bf_t*)(bp + 16),
                                                 0, 1, 2, 3, 4, 5, 6, 7, 8, 9, 10, 11, 12, 13, 14, 15);
            }

            v16bf afrag;
#pragma unroll
            for (int e = 0; e < 4; ++e) {
                afrag[e]      = (__bf16)f0[e];
                afrag[4 + e]  = (__bf16)f1[e];
                afrag[8 + e]  = (__bf16)f2[e];
                afrag[12 + e] = (__bf16)f3[e];
            }

#pragma unroll
            for (int t = 0; t < 7; ++t) {
                acc[t] = __builtin_amdgcn_wmma_f32_16x16x32_bf16(
                    false, afrag, false, bfr[t], (short)0, acc[t], false, false);
            }
        }

        // ---- store: D layout — VGPR r holds row r (lanes 0-15) / 8+r (16-31)
        //      non-temporal: out is write-once, keep x/W resident in L2.
#pragma unroll
        for (int t = 0; t < 7; ++t) {
            const int col  = (g * 7 + t) * 16 + l15;
            const float wv = weight[col];
            float* op = out + (size_t)(rowBase + hi * 8) * NN + col;
#pragma unroll
            for (int r = 0; r < 8; ++r) {
                __builtin_nontemporal_store(acc[t][r] * wv, op + (size_t)r * NN);
            }
        }
    }
}

// ---------------------------------------------------------------------------
// Harness entry. Inputs (setup_inputs order): x, pos2d, weight, ep — all f32.
// d_ws holds the bf16 W matrix (784 * 800 * 2 = 1,254,400 bytes).
// ---------------------------------------------------------------------------
extern "C" void kernel_launch(void* const* d_in, const int* in_sizes, int n_in,
                              void* d_out, int out_size, void* d_ws, size_t ws_size,
                              hipStream_t stream) {
    const float* x      = (const float*)d_in[0];
    const float* pos2d  = (const float*)d_in[1];
    const float* weight = (const float*)d_in[2];
    const float* ep     = (const float*)d_in[3];
    float*       out    = (float*)d_out;
    __bf16*      Wbf    = (__bf16*)d_ws;

    const int wtot = NN * KPAD;
    axs_build_w<<<(wtot + 255) / 256, 256, 0, stream>>>(pos2d, ep, Wbf);
    axs_gemm_wmma<<<BROWS / 128, 256, 0, stream>>>(x, Wbf, weight, out);
}
